// MultiHeadAttention_26688926777396
// MI455X (gfx1250) — matrix operand order repaired and checked by reading, not compiled, not run on hardware
//
#include <hip/hip_runtime.h>

typedef __bf16 bf16_t;
typedef __attribute__((ext_vector_type(16))) __bf16 bf16x16;
typedef __attribute__((ext_vector_type(8)))  __bf16 bf16x8;
typedef __attribute__((ext_vector_type(8)))  float  f32x8;
typedef __attribute__((ext_vector_type(4)))  float  f32x4;

#define B_  4
#define S_  2048
#define DM  1024
#define NH  16
#define DH  64

// ---------------------------------------------------------------- WMMA helpers

static __device__ inline f32x8 wmma_bf16(bf16x16 a, bf16x16 b, f32x8 c) {
  // 8 args: (neg_a, A, neg_b, B, c_mod, C, reuse_a, reuse_b)
  return __builtin_amdgcn_wmma_f32_16x16x32_bf16(false, a, false, b, (short)0, c,
                                                 false, false);
}

static __device__ inline bf16x16 ld_frag2(const bf16_t* p0, const bf16_t* p1) {
  union { bf16x16 v; bf16x8 h[2]; } u;
  u.h[0] = *(const bf16x8*)p0;
  u.h[1] = *(const bf16x8*)p1;
  return u.v;
}

// A-fragment (16xK tile, row-major bf16, row stride ld, K offset kb).
// ISA 16-bit A layout: lanes 0-15: K = kb+0..7 then kb+16..23; lanes 16-31: +8.
static __device__ inline bf16x16 ld_a(const bf16_t* base, int ld, int kb, int lane) {
  int row = lane & 15, sel = lane >> 4;
  const bf16_t* p = base + row * ld + kb + sel * 8;
  return ld_frag2(p, p + 16);
}

// B-fragment from BT (row-major [N][K]): lanes 0-15: K=kb+0..7 then kb+16..23; lanes16-31: +8.
static __device__ inline bf16x16 ld_b(const bf16_t* base, int ld, int kb, int lane) {
  int n = lane & 15, sel = lane >> 4;
  const bf16_t* p = base + n * ld + kb + sel * 8;
  return ld_frag2(p, p + 16);
}

// A-fragment from fp32 row-major, converted to bf16 in registers.
static __device__ inline bf16x16 ld_a_f32(const float* base, int ld, int kb, int lane) {
  int row = lane & 15, sel = lane >> 4;
  const float* p = base + row * ld + kb + sel * 8;
  f32x4 a0 = *(const f32x4*)(p);
  f32x4 a1 = *(const f32x4*)(p + 4);
  f32x4 b0 = *(const f32x4*)(p + 16);
  f32x4 b1 = *(const f32x4*)(p + 20);
  union { bf16x16 v; bf16_t e[16]; } u;
#pragma unroll
  for (int i = 0; i < 4; ++i) {
    u.e[i]      = (bf16_t)a0[i];
    u.e[4 + i]  = (bf16_t)a1[i];
    u.e[8 + i]  = (bf16_t)b0[i];
    u.e[12 + i] = (bf16_t)b1[i];
  }
  return u.v;
}

// 16-lane butterfly reductions (rows of a WMMA C-tile live per half-wave).
static __device__ inline float redmax16(float x) {
#pragma unroll
  for (int m = 1; m <= 8; m <<= 1) x = fmaxf(x, __shfl_xor(x, m, 32));
  return x;
}
static __device__ inline float redsum16(float x) {
#pragma unroll
  for (int m = 1; m <= 8; m <<= 1) x += __shfl_xor(x, m, 32);
  return x;
}

// Async global->LDS copy (ASYNCcnt path, §08).  LDS operand is the byte offset
// within the workgroup LDS allocation = low 32 bits of the generic pointer.
static __device__ inline unsigned lds_off(const void* p) {
  return (unsigned)(uintptr_t)p;
}
#define ASYNC_LDS_B128(ldsoff, gptr)                                        \
  asm volatile("global_load_async_to_lds_b128 %0, %1, off"                  \
               :: "v"(ldsoff),                                              \
                  "v"((unsigned long long)(uintptr_t)(gptr))                \
               : "memory")
#define WAIT_ASYNC(n) asm volatile("s_wait_asynccnt " #n ::: "memory")

// ------------------------------------------------- kernel 0: weight transpose

__global__ __launch_bounds__(256) void k_wprep(const float* __restrict__ wq,
                                               const float* __restrict__ wk,
                                               const float* __restrict__ wv,
                                               const float* __restrict__ wo,
                                               bf16_t* __restrict__ out) {
  const float* src[4] = {wq, wk, wv, wo};
  for (int t = blockIdx.x * 256 + threadIdx.x; t < DM * DM; t += gridDim.x * 256) {
    int o = t >> 10, i = t & (DM - 1);
#pragma unroll
    for (int m = 0; m < 4; ++m)
      out[(size_t)m * DM * DM + t] = (bf16_t)src[m][i * DM + o];  // WT[o][i]=W[i][o]
  }
}

// --------------------------------------------- kernel 1: QKV projection GEMMs
// C[8192x1024] = X_f32(bf16 cvt) @ WT^T + bias. WG: 8 waves -> 64x128 tile.
// mode 0: Q ((acc+b)/8, [b,h,s,d])  mode 1: K ([b,h,s,d])  mode 2: V ([b,h,d,s])

__global__ __launch_bounds__(256) void k_proj(const float* __restrict__ X,
                                              const bf16_t* __restrict__ WT,
                                              const float* __restrict__ bias,
                                              bf16_t* __restrict__ dst, int mode) {
  int lane = threadIdx.x & 31;
  int w = threadIdx.x >> 5;
  int Moff = blockIdx.x * 64 + (w >> 2) * 32;
  int Noff = blockIdx.y * 128 + (w & 3) * 32;
  f32x8 acc[2][2] = {};
  for (int kb = 0; kb < DM; kb += 32) {
    bf16x16 a0 = ld_a_f32(X + (size_t)Moff * DM, DM, kb, lane);
    bf16x16 a1 = ld_a_f32(X + (size_t)(Moff + 16) * DM, DM, kb, lane);
    bf16x16 b0 = ld_b(WT + (size_t)Noff * DM, DM, kb, lane);
    bf16x16 b1 = ld_b(WT + (size_t)(Noff + 16) * DM, DM, kb, lane);
    acc[0][0] = wmma_bf16(a0, b0, acc[0][0]);
    acc[0][1] = wmma_bf16(a0, b1, acc[0][1]);
    acc[1][0] = wmma_bf16(a1, b0, acc[1][0]);
    acc[1][1] = wmma_bf16(a1, b1, acc[1][1]);
  }
  int c0 = lane & 15, sel = lane >> 4;
  float scale = (mode == 0) ? 0.125f : 1.0f;  // 1/sqrt(D_K) folded into Q
#pragma unroll
  for (int mt = 0; mt < 2; ++mt) {
#pragma unroll
    for (int nt = 0; nt < 2; ++nt) {
      int n = Noff + nt * 16 + c0;
      float bv = bias[n];
      int h = n >> 6, d = n & 63;
      const float* a = (const float*)&acc[mt][nt];
#pragma unroll
      for (int i = 0; i < 8; ++i) {
        int row = Moff + mt * 16 + i + 8 * sel;
        int b = row >> 11, s = row & (S_ - 1);
        float v = (a[i] + bv) * scale;
        size_t idx = (mode == 2)
            ? ((size_t)((b * NH + h) * DH + d) * S_ + s)    // V^T: [b,h,d,s]
            : ((size_t)((b * NH + h) * S_ + s) * DH + d);   // Q,K: [b,h,s,d]
        dst[idx] = (bf16_t)v;
      }
    }
  }
}

// --------------------------------------- kernel 2: flash attention (bf16 WMMA)
// One wave = 16 queries, 8 waves/WG share (b,h).  K/V 32-key tiles are staged
// into LDS once per WG via double-buffered async copies (ASYNCcnt) so the next
// tile's DMA overlaps the current tile's WMMAs + online softmax.

__global__ __launch_bounds__(256) void k_attn(const bf16_t* __restrict__ Qs,
                                              const bf16_t* __restrict__ Ks,
                                              const bf16_t* __restrict__ VT,
                                              bf16_t* __restrict__ AO) {
  __shared__ __align__(16) bf16_t Kt[2][32 * DH];   // [key][64] per buffer
  __shared__ __align__(16) bf16_t Vt[2][DH * 32];   // [feat][32] per buffer
  __shared__ __align__(16) bf16_t Pbuf[8][16 * 32]; // per-wave P (C->A layout)
  int tid = threadIdx.x;
  int lane = tid & 31, w = tid >> 5;
  int b = blockIdx.z, h = blockIdx.y;
  int qb = blockIdx.x * 128 + w * 16;
  const bf16_t* Qbh = Qs + (size_t)(b * NH + h) * S_ * DH;
  const bf16_t* Kbh = Ks + (size_t)(b * NH + h) * S_ * DH;
  const bf16_t* Vbh = VT + (size_t)(b * NH + h) * DH * S_;
  int c0 = lane & 15, sel = lane >> 4;

  bf16x16 qf0 = ld_a(Qbh + (size_t)qb * DH, DH, 0, lane);   // K-dim 0..31
  bf16x16 qf1 = ld_a(Qbh + (size_t)qb * DH, DH, 32, lane);  // K-dim 32..63

  // Per-thread async-copy slots: K tile = 256 x 16B, V tile = 64 rows x 4 x 16B.
  const bf16_t* gK = Kbh + tid * 8;                                  // +jb*DH/step
  const bf16_t* gV = Vbh + (size_t)(tid >> 2) * S_ + (tid & 3) * 8;  // +jb/step
  unsigned lK[2] = { lds_off(&Kt[0][tid * 8]),  lds_off(&Kt[1][tid * 8]) };
  unsigned lV[2] = { lds_off(&Vt[0][(tid >> 2) * 32 + (tid & 3) * 8]),
                     lds_off(&Vt[1][(tid >> 2) * 32 + (tid & 3) * 8]) };

  f32x8 O[4] = {};
  float rm[8], rl[8];
#pragma unroll
  for (int i = 0; i < 8; ++i) { rm[i] = -1e30f; rl[i] = 0.f; }

  // prologue: start DMA of first tile
  ASYNC_LDS_B128(lK[0], gK);
  ASYNC_LDS_B128(lV[0], gV);

  int cur = 0;
  for (int jb = 0; jb < S_; jb += 32) {
    int nxt = cur ^ 1;
    if (jb + 32 < S_) {  // prefetch next tile, then retire current (in-order)
      ASYNC_LDS_B128(lK[nxt], gK + (size_t)(jb + 32) * DH);
      ASYNC_LDS_B128(lV[nxt], gV + (jb + 32));
      WAIT_ASYNC(0x2);
    } else {
      WAIT_ASYNC(0x0);
    }
    __syncthreads();  // current tile visible to all waves

    f32x8 S0 = {}, S1 = {};
    {
      bf16x16 bk;
      bk = ld_b(&Kt[cur][0], DH, 0, lane);          S0 = wmma_bf16(qf0, bk, S0);
      bk = ld_b(&Kt[cur][0], DH, 32, lane);         S0 = wmma_bf16(qf1, bk, S0);
      bk = ld_b(&Kt[cur][16 * DH], DH, 0, lane);    S1 = wmma_bf16(qf0, bk, S1);
      bk = ld_b(&Kt[cur][16 * DH], DH, 32, lane);   S1 = wmma_bf16(qf1, bk, S1);
    }
    const float* s0 = (const float*)&S0;
    const float* s1 = (const float*)&S1;
#pragma unroll
    for (int i = 0; i < 8; ++i) {
      float mx = redmax16(fmaxf(s0[i], s1[i]));
      float nm = fmaxf(rm[i], mx);
      float sc = __expf(rm[i] - nm);
      rm[i] = nm;
      float p0 = __expf(s0[i] - nm);
      float p1 = __expf(s1[i] - nm);
      rl[i] = rl[i] * sc + redsum16(p0 + p1);
#pragma unroll
      for (int nt = 0; nt < 4; ++nt) ((float*)&O[nt])[i] *= sc;
      int row = i + 8 * sel;
      Pbuf[w][row * 32 + c0]      = (bf16_t)p0;  // keys jb..jb+15
      Pbuf[w][row * 32 + 16 + c0] = (bf16_t)p1;  // keys jb+16..jb+31
    }
    asm volatile("s_wait_dscnt 0x0" ::: "memory");  // P store -> cross-lane reload
    bf16x16 pf = ld_a(&Pbuf[w][0], 32, 0, lane);
#pragma unroll
    for (int nt = 0; nt < 4; ++nt) {
      bf16x16 bv = ld_b(&Vt[cur][nt * 16 * 32], 32, 0, lane);
      O[nt] = wmma_bf16(pf, bv, O[nt]);
    }
    __syncthreads();  // everyone done reading cur before it is re-filled
    cur = nxt;
  }

#pragma unroll
  for (int nt = 0; nt < 4; ++nt) {
    const float* o = (const float*)&O[nt];
#pragma unroll
    for (int i = 0; i < 8; ++i) {
      int row = qb + i + 8 * sel;
      int col = h * DH + nt * 16 + c0;
      AO[(size_t)(b * S_ + row) * DM + col] = (bf16_t)(o[i] / rl[i]);
    }
  }
}

// ----------------------------------- kernel 3: out-projection + bias + LayerNorm
// WG owns 16 full rows (8 waves x 128 cols) so mean/var reduce in-block.

__global__ __launch_bounds__(256) void k_oproj_ln(const bf16_t* __restrict__ AO,
                                                  const bf16_t* __restrict__ WOT,
                                                  const float* __restrict__ b_o,
                                                  const float* __restrict__ ln_g,
                                                  const float* __restrict__ ln_b,
                                                  float* __restrict__ out) {
  __shared__ float red[2][8][16];  // [sum|sumsq][wave][row]
  int lane = threadIdx.x & 31, w = threadIdx.x >> 5;
  int Moff = blockIdx.x * 16;
  int Noff = w * 128;
  f32x8 acc[8] = {};
  for (int kb = 0; kb < DM; kb += 32) {
    bf16x16 a = ld_a(AO + (size_t)Moff * DM, DM, kb, lane);
#pragma unroll
    for (int nt = 0; nt < 8; ++nt) {
      bf16x16 bw = ld_b(WOT + (size_t)(Noff + nt * 16) * DM, DM, kb, lane);
      acc[nt] = wmma_bf16(a, bw, acc[nt]);
    }
  }
  int c0 = lane & 15, sel = lane >> 4;
  float ps[8], pq[8];
#pragma unroll
  for (int i = 0; i < 8; ++i) { ps[i] = 0.f; pq[i] = 0.f; }
#pragma unroll
  for (int nt = 0; nt < 8; ++nt) {
    float bv = b_o[Noff + nt * 16 + c0];
    float* a = (float*)&acc[nt];
#pragma unroll
    for (int i = 0; i < 8; ++i) {
      a[i] += bv;
      ps[i] += a[i];
      pq[i] += a[i] * a[i];
    }
  }
#pragma unroll
  for (int i = 0; i < 8; ++i) { ps[i] = redsum16(ps[i]); pq[i] = redsum16(pq[i]); }
  if (c0 == 0) {
#pragma unroll
    for (int i = 0; i < 8; ++i) {
      red[0][w][i + 8 * sel] = ps[i];
      red[1][w][i + 8 * sel] = pq[i];
    }
  }
  __syncthreads();
#pragma unroll
  for (int i = 0; i < 8; ++i) {
    int row = i + 8 * sel;
    float s = 0.f, q = 0.f;
#pragma unroll
    for (int w2 = 0; w2 < 8; ++w2) { s += red[0][w2][row]; q += red[1][w2][row]; }
    float mean = s * (1.0f / DM);
    float var = q * (1.0f / DM) - mean * mean;
    float rinv = rsqrtf(var + 1e-6f);
#pragma unroll
    for (int nt = 0; nt < 8; ++nt) {
      int col = Noff + nt * 16 + c0;
      float v = ((float*)&acc[nt])[i];
      out[(size_t)(Moff + row) * DM + col] = (v - mean) * rinv * ln_g[col] + ln_b[col];
    }
  }
}

// --------------------------------------------------------------------- launch

extern "C" void kernel_launch(void* const* d_in, const int* in_sizes, int n_in,
                              void* d_out, int out_size, void* d_ws, size_t ws_size,
                              hipStream_t stream) {
  (void)in_sizes; (void)n_in; (void)out_size; (void)ws_size;
  const float* q    = (const float*)d_in[0];
  const float* k    = (const float*)d_in[1];
  const float* v    = (const float*)d_in[2];
  const float* w_q  = (const float*)d_in[3];
  const float* b_q  = (const float*)d_in[4];
  const float* w_k  = (const float*)d_in[5];
  const float* b_k  = (const float*)d_in[6];
  const float* w_v  = (const float*)d_in[7];
  const float* b_v  = (const float*)d_in[8];
  const float* w_o  = (const float*)d_in[9];
  const float* b_o  = (const float*)d_in[10];
  const float* ln_g = (const float*)d_in[11];
  const float* ln_b = (const float*)d_in[12];

  const size_t WSZ = (size_t)DM * DM;                 // 1M elems per weight
  const size_t TSZ = (size_t)B_ * NH * S_ * DH;       // 8M elems per tensor
  bf16_t* ws  = (bf16_t*)d_ws;
  bf16_t* WQT = ws;                                   // [out][in] bf16
  bf16_t* WKT = WQT + WSZ;
  bf16_t* WVT = WKT + WSZ;
  bf16_t* WOT = WVT + WSZ;
  bf16_t* Qs  = WOT + WSZ;                            // [b,h,s,d] (pre-scaled)
  bf16_t* Ks  = Qs + TSZ;                             // [b,h,s,d]
  bf16_t* VTt = Ks + TSZ;                             // [b,h,d,s]
  bf16_t* AO  = VTt + TSZ;                            // [b*s][h*d]

  k_wprep<<<1024, 256, 0, stream>>>(w_q, w_k, w_v, w_o, WQT);

  dim3 gp(128, 8);  // 8192/64 x 1024/128
  k_proj<<<gp, 256, 0, stream>>>(q, WQT, b_q, Qs, 0);
  k_proj<<<gp, 256, 0, stream>>>(k, WKT, b_k, Ks, 1);
  k_proj<<<gp, 256, 0, stream>>>(v, WVT, b_v, VTt, 2);

  dim3 ga(S_ / 128, NH, B_);
  k_attn<<<ga, 256, 0, stream>>>(Qs, Ks, VTt, AO);

  k_oproj_ln<<<(B_ * S_) / 16, 256, 0, stream>>>(AO, WOT, b_o, ln_g, ln_b,
                                                 (float*)d_out);
}
